// RSSMRNNLayer_29532195127352
// MI455X (gfx1250) — matrix operand order repaired and checked by reading, not compiled, not run on hardware
//
#include <hip/hip_runtime.h>

typedef __attribute__((ext_vector_type(16))) __bf16 v16bf;
typedef __attribute__((ext_vector_type(8)))  float  v8f;

struct U4x2 { uint4 lo; uint4 hi; };

__device__ __forceinline__ unsigned short f2bfu(float f) {
    __bf16 h = (__bf16)f;                      // native cvt, RNE
    return __builtin_bit_cast(unsigned short, h);
}

__device__ __forceinline__ v16bf load_frag(const unsigned short* p0,
                                           const unsigned short* p1) {
    U4x2 t;
    t.lo = *(const uint4*)p0;                  // ds_load_b128
    t.hi = *(const uint4*)p1;                  // ds_load_b128
    return __builtin_bit_cast(v16bf, t);
}

// Truncate flat shared-memory address to LDS byte offset (low 32 bits of aperture).
__device__ __forceinline__ unsigned int lds_off(const void* p) {
    return (unsigned int)(uintptr_t)p;
}

// Async DMA one 16B chunk global -> LDS (ASYNCcnt path).
__device__ __forceinline__ void async_copy_b128(unsigned int ldsoff, const void* gaddr) {
    asm volatile("global_load_async_to_lds_b128 %0, %1, off"
                 :: "v"(ldsoff), "v"(gaddr) : "memory");
}

// ---------------- pre-pass 1: fp32 -> bf16 rows (optionally reset-masked) -------------
__global__ __launch_bounds__(256)
void convert_rows_kernel(const float* __restrict__ src,
                         const int*   __restrict__ reset,
                         unsigned short* __restrict__ dst,
                         int applyReset)
{
    size_t idx8 = (size_t)blockIdx.x * 256 + threadIdx.x;   // 8 elems per thread
    size_t base = idx8 * 8;                                  // row = base >> 10 (D=U=1024)
    float4 f0 = *(const float4*)(src + base);
    float4 f1 = *(const float4*)(src + base + 4);
    if (applyReset && reset[base >> 10]) {
        f0.x = f0.y = f0.z = f0.w = 0.f;
        f1.x = f1.y = f1.z = f1.w = 0.f;
    }
    uint4 u;
    u.x = (unsigned)f2bfu(f0.x) | ((unsigned)f2bfu(f0.y) << 16);
    u.y = (unsigned)f2bfu(f0.z) | ((unsigned)f2bfu(f0.w) << 16);
    u.z = (unsigned)f2bfu(f1.x) | ((unsigned)f2bfu(f1.y) << 16);
    u.w = (unsigned)f2bfu(f1.z) | ((unsigned)f2bfu(f1.w) << 16);
    *(uint4*)(dst + base) = u;
}

// ---------------- pre-pass 2: W (1024 x 3072 f32, row-major) -> W^T (3072 x 1024 bf16) -
__global__ __launch_bounds__(256)
void transpose_w_kernel(const float* __restrict__ w, unsigned short* __restrict__ wT)
{
    __shared__ __align__(16) unsigned short lt[32 * 36];   // 32x32 tile, padded stride
    const int tid = threadIdx.x;
    const int c0  = blockIdx.x * 32;   // column tile (0..3071)
    const int k0  = blockIdx.y * 32;   // k tile (0..1023)

    {   // coalesced read, transposed scatter into LDS
        int r = tid >> 3, q = tid & 7;
        float4 f = *(const float4*)(w + (size_t)(k0 + r) * 3072 + c0 + q * 4);
        lt[(q * 4 + 0) * 36 + r] = f2bfu(f.x);
        lt[(q * 4 + 1) * 36 + r] = f2bfu(f.y);
        lt[(q * 4 + 2) * 36 + r] = f2bfu(f.z);
        lt[(q * 4 + 3) * 36 + r] = f2bfu(f.w);
    }
    __syncthreads();
    {   // coalesced write of transposed tile
        int cc = tid >> 3, q2 = tid & 7;
        uint2 u = *(const uint2*)&lt[cc * 36 + q2 * 4];
        *(uint2*)(wT + (size_t)(c0 + cc) * 1024 + k0 + q2 * 4) = u;
    }
}

// ---------------- fused GEMM + GRU epilogue, double-buffered async staging ------------
// Workgroup: 64 rows x 64 j-cols; planes z, r, a_x, a_h.
__global__ __launch_bounds__(256)
void rssm_gru_fused_kernel(const unsigned short* __restrict__ xb,
                           const unsigned short* __restrict__ hb,
                           const unsigned short* __restrict__ wiT,
                           const unsigned short* __restrict__ whT,
                           const float* __restrict__ b,
                           const int*   __restrict__ reset,
                           const float* __restrict__ state,
                           float* __restrict__ out)
{
    // double-buffered tiles
    __shared__ __align__(16) unsigned short lA[2][64 * 40];       // 64 rows x 32 k
    __shared__ __align__(16) unsigned short lB[2][3 * 64 * 40];   // 192 cols x 32 k
    const unsigned int LA_STRIDE = 64 * 40 * 2;        // bytes between buffers
    const unsigned int LB_STRIDE = 3 * 64 * 40 * 2;

    const int tid   = threadIdx.x;
    const int lane  = tid & 31;
    const int wave  = tid >> 5;
    const int waveM = wave & 3;
    const int waveN = wave >> 2;
    const int rowl  = lane & 15;
    const int halfl = lane >> 4;

    const int m0 = blockIdx.y * 64;
    const int c0 = blockIdx.x * 64;

    // Fixed per-thread staging assignment (contiguous 16B chunks).
    const int arow = tid >> 2, ach = tid & 3;                       // A: 256 chunks
    const unsigned int ldsA0 = lds_off(&lA[0][arow * 40 + ach * 8]);
    unsigned int ldsB0[3];
    int bgate[3], bcolc[3], bch[3];
    #pragma unroll
    for (int it = 0; it < 3; ++it) {
        int q = tid + it * 256;            // 0..767
        int cidx = q >> 2;                 // 0..191  (gate*64 + col)
        bch[it]   = q & 3;
        bgate[it] = cidx >> 6;
        bcolc[it] = cidx & 63;
        ldsB0[it] = lds_off(&lB[0][cidx * 40 + bch[it] * 8]);
    }

    // Start-of-phase0 global pointers; phase1 regions sit at fixed offsets
    // (hb = xb + 8M halves, whT = wiT + 3M halves, identical layouts).
    const unsigned short* gaS = xb + (size_t)(m0 + arow) * 1024 + ach * 8;
    const unsigned short* gbS[3];
    #pragma unroll
    for (int it = 0; it < 3; ++it)
        gbS[it] = wiT + (size_t)(bgate[it] * 1024 + c0 + bcolc[it]) * 1024 + bch[it] * 8;

    const unsigned short* ga  = gaS;
    const unsigned short* gb0 = gbS[0];
    const unsigned short* gb1 = gbS[1];
    const unsigned short* gb2 = gbS[2];

    // prologue: DMA step 0 into buffer 0
    async_copy_b128(ldsA0,    ga);
    async_copy_b128(ldsB0[0], gb0);
    async_copy_b128(ldsB0[1], gb1);
    async_copy_b128(ldsB0[2], gb2);

    v8f acc[4][2] = {};   // 0=z, 1=r, 2=a_x, 3=a_h

    for (int s = 0; s < 64; ++s) {            // 64 K-steps: 32 x-phase, 32 h-phase
        __syncthreads();                       // all waves done reading buf[(s+1)&1]
        if (s + 1 < 64) {
            if (s + 1 == 32) {                 // phase boundary: jump to hb / whT
                ga  = gaS    + (size_t)8192 * 1024;
                gb0 = gbS[0] + (size_t)3072 * 1024;
                gb1 = gbS[1] + (size_t)3072 * 1024;
                gb2 = gbS[2] + (size_t)3072 * 1024;
            } else {
                ga += 32; gb0 += 32; gb1 += 32; gb2 += 32;
            }
            unsigned int bo = ((unsigned)(s + 1) & 1u);
            async_copy_b128(ldsA0    + bo * LA_STRIDE, ga);
            async_copy_b128(ldsB0[0] + bo * LB_STRIDE, gb0);
            async_copy_b128(ldsB0[1] + bo * LB_STRIDE, gb1);
            async_copy_b128(ldsB0[2] + bo * LB_STRIDE, gb2);
            __builtin_prefetch((const void*)(ga + 32), 0, 3);
            // older 4 DMAs (this step's tile) done; newer 4 stay in flight
            asm volatile("s_wait_asynccnt 4" ::: "memory");
        } else {
            asm volatile("s_wait_asynccnt 0" ::: "memory");
        }
        __syncthreads();                       // every wave's chunks landed

        const int buf   = s & 1;
        const int phase = s >> 5;

        // ---- A fragment (16x32 bf16 layout) ----
        const unsigned short* pa = &lA[buf][(waveM * 16 + rowl) * 40 + halfl * 16];
        v16bf afrag = load_frag(pa, pa + 8);

        #pragma unroll
        for (int g = 0; g < 2; ++g) {            // z, r planes
            #pragma unroll
            for (int cg = 0; cg < 2; ++cg) {
                int col = waveN * 32 + cg * 16 + rowl;
                const unsigned short* pb = &lB[buf][(g * 64 + col) * 40 + halfl * 16];
                v16bf bfrag = load_frag(pb, pb + 8);
                acc[g][cg] = __builtin_amdgcn_wmma_f32_16x16x32_bf16(
                    false, afrag, false, bfrag, (short)0, acc[g][cg], false, false);
            }
        }
        if (phase == 0) {                        // a_x plane
            #pragma unroll
            for (int cg = 0; cg < 2; ++cg) {
                int col = waveN * 32 + cg * 16 + rowl;
                const unsigned short* pb = &lB[buf][(2 * 64 + col) * 40 + halfl * 16];
                v16bf bfrag = load_frag(pb, pb + 8);
                acc[2][cg] = __builtin_amdgcn_wmma_f32_16x16x32_bf16(
                    false, afrag, false, bfrag, (short)0, acc[2][cg], false, false);
            }
        } else {                                 // a_h plane (kept separate for r*)
            #pragma unroll
            for (int cg = 0; cg < 2; ++cg) {
                int col = waveN * 32 + cg * 16 + rowl;
                const unsigned short* pb = &lB[buf][(2 * 64 + col) * 40 + halfl * 16];
                v16bf bfrag = load_frag(pb, pb + 8);
                acc[3][cg] = __builtin_amdgcn_wmma_f32_16x16x32_bf16(
                    false, afrag, false, bfrag, (short)0, acc[3][cg], false, false);
            }
        }
    }

    // ---- GRU epilogue: C-layout row = v + 8*(lane/16), col = lane%16 ----
    const size_t OUT2 = (size_t)8192 * 1024;
    #pragma unroll
    for (int cg = 0; cg < 2; ++cg) {
        int colU = c0 + waveN * 32 + cg * 16 + rowl;
        float bz = b[colU];
        float br = b[1024 + colU];
        float ba = b[2048 + colU];
        #pragma unroll
        for (int v = 0; v < 8; ++v) {
            int row = m0 + waveM * 16 + halfl * 8 + v;
            size_t idx = (size_t)row * 1024 + colU;
            float hv = reset[row] ? 0.f : state[idx];
            float z = 1.f / (1.f + __expf(-(acc[0][cg][v] + bz)));
            float r = 1.f / (1.f + __expf(-(acc[1][cg][v] + br)));
            float a = tanhf(acc[2][cg][v] + ba + r * acc[3][cg][v]);
            float hn = (1.f - z) * hv + z * a;
            out[idx]        = hn;
            out[OUT2 + idx] = hn;
        }
    }
}

extern "C" void kernel_launch(void* const* d_in, const int* in_sizes, int n_in,
                              void* d_out, int out_size, void* d_ws, size_t ws_size,
                              hipStream_t stream) {
    const float* x     = (const float*)d_in[0];
    const int*   reset = (const int*)  d_in[1];
    const float* state = (const float*)d_in[2];
    const float* w_i   = (const float*)d_in[3];
    const float* w_h   = (const float*)d_in[4];
    const float* b     = (const float*)d_in[5];
    float* out = (float*)d_out;

    char* ws = (char*)d_ws;
    unsigned short* xb  = (unsigned short*)(ws);                          // 16 MB
    unsigned short* hb  = (unsigned short*)(ws + ((size_t)16 << 20));     // 16 MB (xb + 8M halves)
    unsigned short* wiT = (unsigned short*)(ws + ((size_t)32 << 20));     //  6 MB
    unsigned short* whT = (unsigned short*)(ws + ((size_t)38 << 20));     //  6 MB (wiT + 3M halves)

    // pre-passes: convert / mask / transpose to bf16 once
    convert_rows_kernel<<<4096, 256, 0, stream>>>(x,     reset, xb, 0);
    convert_rows_kernel<<<4096, 256, 0, stream>>>(state, reset, hb, 1);
    transpose_w_kernel<<<dim3(3072 / 32, 1024 / 32), 256, 0, stream>>>(w_i, wiT);
    transpose_w_kernel<<<dim3(3072 / 32, 1024 / 32), 256, 0, stream>>>(w_h, whT);

    // fused GEMM + GRU
    dim3 grid(1024 / 64, 8192 / 64);
    rssm_gru_fused_kernel<<<grid, 256, 0, stream>>>(xb, hb, wiT, whT, b, reset, state, out);
}
